// GPT2Block_1700807049337
// MI455X (gfx1250) — matrix operand order repaired and checked by reading, not compiled, not run on hardware
//
#include <hip/hip_runtime.h>
#include <hip/hip_bf16.h>

// GPT-2 block forward for gfx1250 (MI455X): bf16 WMMA + async global->LDS
// double-buffered staging. B=4, T=2048, C=768, H=12, D=64.

#define Bsz  4
#define Tseq 2048
#define Cdim 768
#define Hh   12
#define Dh   64
#define Mrows (Bsz*Tseq)            // 8192
#define BHTD  (Bsz*Hh*Tseq*Dh)      // per-tensor packed q/k/v elements

typedef __attribute__((ext_vector_type(16))) __bf16       bf16x16;
typedef __attribute__((ext_vector_type(8)))  float        f32x8;

__device__ __forceinline__ f32x8 wmma_bf16(bf16x16 a, bf16x16 b, f32x8 c) {
  return __builtin_amdgcn_wmma_f32_16x16x32_bf16(false, a, false, b,
                                                 (short)0, c, false, false);
}

// Async copy 16B global -> LDS, tracked by ASYNCcnt.
// lds_off: byte offset in LDS (low 32 bits of generic shared pointer).
__device__ __forceinline__ void async_b128(unsigned lds_off, unsigned long long g) {
  asm volatile("global_load_async_to_lds_b128 %0, %1, off"
               :: "v"(lds_off), "v"(g) : "memory");
}
__device__ __forceinline__ unsigned lds_off(const void* p) {
  return (unsigned)(unsigned long long)p;   // LDS aperture: addr[31:0] = LDS offset
}

// A-fragment (16x32 bf16). LDS tile row-major [m][k], row stride ld (elems).
// Lane l: m = l&15, g = l>>4; VGPR v<4: K = 8g+2v, V4..7: K = 16+8g+2(v-4).
__device__ __forceinline__ bf16x16 frag_a(const __bf16* base, int ld, int lane) {
  int m = lane & 15, g = lane >> 4;
  union { bf16x16 v; unsigned int u[8]; } r;
  const __bf16* p = base + m * ld + 8 * g;
#pragma unroll
  for (int i = 0; i < 4; ++i) r.u[i]     = *(const unsigned int*)(p + 2*i);
#pragma unroll
  for (int i = 0; i < 4; ++i) r.u[4 + i] = *(const unsigned int*)(p + 16 + 2*i);
  return r.v;
}

// B-fragment (32x16 bf16, KxN). LDS tile stored transposed [n][k], stride ld.
// Lane l: n = l&15, g = l>>4; VGPR v<4: K = 8g+2v, V4..7: K = 16+8g+2(v-4).
__device__ __forceinline__ bf16x16 frag_b(const __bf16* base, int ld, int lane) {
  int n = lane & 15, g = lane >> 4;
  union { bf16x16 v; unsigned int u[8]; } r;
  const __bf16* p = base + n * ld + 8 * g;
#pragma unroll
  for (int i = 0; i < 8; ++i) r.u[i] = *(const unsigned int*)(p + 2*i + (i >= 4 ? 8 : 0));
  return r.v;
}

// ------------------------------------------------ transpose-convert weights
// in: [K][N] f32  ->  out: [N][K] bf16  (so GEMM B tiles need no LDS transpose)
__global__ __launch_bounds__(256) void cvt_transpose_kernel(
    const float* __restrict__ in, __bf16* __restrict__ out, int K, int N) {
  __shared__ float tile[32][33];
  int k0 = blockIdx.y * 32, n0 = blockIdx.x * 32;
  int tx = threadIdx.x & 31, ty = threadIdx.x >> 5;   // 32 x 8
#pragma unroll
  for (int i = ty; i < 32; i += 8)
    tile[i][tx] = in[(size_t)(k0 + i) * N + n0 + tx];
  __syncthreads();
#pragma unroll
  for (int i = ty; i < 32; i += 8)
    out[(size_t)(n0 + i) * K + k0 + tx] = (__bf16)tile[tx][i];
}

// ---------------------------------------------------------------- layernorm
__global__ __launch_bounds__(256) void ln_kernel(
    const float* __restrict__ x, const float* __restrict__ g,
    const float* __restrict__ b, __bf16* __restrict__ out) {
  int row = blockIdx.x;
  const float* xr = x + (size_t)row * Cdim;
  float s = 0.f, s2 = 0.f;
  for (int i = threadIdx.x; i < Cdim; i += 256) { float v = xr[i]; s += v; s2 += v*v; }
#pragma unroll
  for (int off = 16; off > 0; off >>= 1) {
    s  += __shfl_xor(s,  off, 32);
    s2 += __shfl_xor(s2, off, 32);
  }
  __shared__ float as[8], as2[8];
  int wv = threadIdx.x >> 5, ln = threadIdx.x & 31;
  if (ln == 0) { as[wv] = s; as2[wv] = s2; }
  __syncthreads();
  if (wv == 0) {
    float t1 = (ln < 8) ? as[ln]  : 0.f;
    float t2 = (ln < 8) ? as2[ln] : 0.f;
#pragma unroll
    for (int off = 4; off > 0; off >>= 1) {
      t1 += __shfl_xor(t1, off, 32);
      t2 += __shfl_xor(t2, off, 32);
    }
    if (ln == 0) { as[0] = t1; as2[0] = t2; }
  }
  __syncthreads();
  float mu   = as[0] * (1.f / Cdim);
  float var  = as2[0] * (1.f / Cdim) - mu * mu;
  float rstd = rsqrtf(var + 1e-5f);
  __bf16* orow = out + (size_t)row * Cdim;
  for (int i = threadIdx.x; i < Cdim; i += 256)
    orow[i] = (__bf16)((xr[i] - mu) * rstd * g[i] + b[i]);
}

// ---------------------------------------------------------------- GEMM
// C[M,N] = A[M,K] * Wt[N,K]^T; double-buffered async LDS staging.
#define LDT 40   // padded LDS stride: 16B-aligned rows, conflict-free frags
enum { MODE_QKV = 0, MODE_RES = 1, MODE_GELU = 2 };

template<int MODE>
__global__ __launch_bounds__(256) void gemm_bf16_kernel(
    const __bf16* __restrict__ A, const __bf16* __restrict__ Wt,
    const float* __restrict__ bias, const float* __restrict__ res,
    float* __restrict__ outF, __bf16* __restrict__ outB, int N, int K) {
  __shared__ __bf16 sA[2][128 * LDT];
  __shared__ __bf16 sB[2][128 * LDT];   // [n][k]
  const int bm = blockIdx.y * 128, bn = blockIdx.x * 128;
  const int t = threadIdx.x, lane = t & 31, wv = t >> 5;
  const int wm = wv & 3, wn = wv >> 2;  // wave tile: 32 rows x 64 cols

  auto stage = [&](int buf, int k0) {   // 4 async b128 per thread
#pragma unroll
    for (int j = 0; j < 2; ++j) {
      int c = t + 256 * j;              // 0..511
      int r = c >> 2, ko = (c & 3) * 8;
      async_b128(lds_off(&sA[buf][r * LDT + ko]),
                 (unsigned long long)(A + (size_t)(bm + r) * K + k0 + ko));
    }
#pragma unroll
    for (int j = 0; j < 2; ++j) {
      int c = t + 256 * j;
      int r = c >> 2, ko = (c & 3) * 8;
      async_b128(lds_off(&sB[buf][r * LDT + ko]),
                 (unsigned long long)(Wt + (size_t)(bn + r) * K + k0 + ko));
    }
  };

  f32x8 acc[2][4];
#pragma unroll
  for (int mt = 0; mt < 2; ++mt)
#pragma unroll
    for (int nt = 0; nt < 4; ++nt)
#pragma unroll
      for (int i = 0; i < 8; ++i) acc[mt][nt][i] = 0.f;

  const int nk = K / 32;
  stage(0, 0);
  for (int ki = 0; ki < nk; ++ki) {
    int cur = ki & 1;
    if (ki + 1 < nk) {
      stage(cur ^ 1, (ki + 1) * 32);
      asm volatile("s_wait_asynccnt 0x4" ::: "memory");  // batch ki resident
    } else {
      asm volatile("s_wait_asynccnt 0x0" ::: "memory");
    }
    __syncthreads();
    const __bf16* Ab = sA[cur];
    const __bf16* Bb = sB[cur];
    bf16x16 a0 = frag_a(Ab + (wm * 32) * LDT, LDT, lane);
    bf16x16 a1 = frag_a(Ab + (wm * 32 + 16) * LDT, LDT, lane);
#pragma unroll
    for (int nt = 0; nt < 4; ++nt) {
      bf16x16 bfr = frag_b(Bb + (wn * 64 + nt * 16) * LDT, LDT, lane);
      acc[0][nt] = wmma_bf16(a0, bfr, acc[0][nt]);
      acc[1][nt] = wmma_bf16(a1, bfr, acc[1][nt]);
    }
    __syncthreads();   // done reading buf[cur] before it is restaged
  }

  const int n = lane & 15, g = lane >> 4;
#pragma unroll
  for (int mt = 0; mt < 2; ++mt)
#pragma unroll
    for (int nt = 0; nt < 4; ++nt)
#pragma unroll
      for (int v = 0; v < 8; ++v) {
        int row = bm + wm * 32 + mt * 16 + v + 8 * g;
        int col = bn + wn * 64 + nt * 16 + n;
        float val = acc[mt][nt][v] + bias[col];
        if constexpr (MODE == MODE_QKV) {
          int sect = col / Cdim, cc = col - sect * Cdim;
          int hh = cc >> 6, d = cc & 63;
          int bb = row >> 11, tt = row & (Tseq - 1);
          size_t dst;
          if (sect == 0) {            // Q: [B,H,T,D], 1/sqrt(D) folded in
            val *= 0.125f;
            dst = (((size_t)bb * Hh + hh) * Tseq + tt) * Dh + d;
          } else if (sect == 1) {     // K: [B,H,T,D]
            dst = (size_t)BHTD + (((size_t)bb * Hh + hh) * Tseq + tt) * Dh + d;
          } else {                    // V: [B,H,D,T]  (pre-transposed for PV)
            dst = 2 * (size_t)BHTD + (((size_t)bb * Hh + hh) * Dh + d) * Tseq + tt;
          }
          outB[dst] = (__bf16)val;
        } else if constexpr (MODE == MODE_RES) {
          size_t idx = (size_t)row * N + col;
          outF[idx] = val + res[idx];
        } else {  // bias + tanh-GELU -> bf16
          float u  = val;
          float th = tanhf(0.7978845608028654f * (u + 0.044715f * u * u * u));
          outB[(size_t)row * N + col] = (__bf16)(0.5f * u * (1.f + th));
        }
      }
}

// ---------------------------------------------------------------- flash attn
#define LDF 72   // padded LDS stride for 64-wide tiles (16B-aligned rows)
__global__ __launch_bounds__(256) void flash_kernel(
    const __bf16* __restrict__ qp, const __bf16* __restrict__ kp,
    const __bf16* __restrict__ vp, __bf16* __restrict__ y) {
  __shared__ __bf16 qs[128 * LDF];       // Q tile [q][d]
  __shared__ __bf16 ks[2][64 * LDF];     // K tile [t][d]
  __shared__ __bf16 vst[2][64 * LDF];    // V tile [d][t] (packed transposed)
  __shared__ __bf16 ps[8 * 16 * LDF];    // per-wave P scratch [m][t]
  const int bh = blockIdx.x, qt = blockIdx.y;
  const int bb = bh / Hh, hh = bh % Hh;
  const __bf16* qh = qp + (size_t)bh * Tseq * Dh;
  const __bf16* kh = kp + (size_t)bh * Tseq * Dh;
  const __bf16* vh = vp + (size_t)bh * Dh * Tseq;
  const int qbase = qt * 128;
  const int t = threadIdx.x, lane = t & 31, wv = t >> 5;
  const int nn = lane & 15, gg = lane >> 4;

  {   // Q tile 128x64: 4 async b128 per thread
#pragma unroll
    for (int j = 0; j < 4; ++j) {
      int c = t + 256 * j;               // 0..1023
      int r = c >> 3, dof = (c & 7) * 8;
      async_b128(lds_off(&qs[r * LDF + dof]),
                 (unsigned long long)(qh + (size_t)(qbase + r) * Dh + dof));
    }
  }
  auto stage_kv = [&](int buf, int kt) {  // 4 async b128 per thread
    int kbase = kt * 64;
#pragma unroll
    for (int j = 0; j < 2; ++j) {
      int c = t + 256 * j;               // 0..511
      int r = c >> 3, dof = (c & 7) * 8;
      async_b128(lds_off(&ks[buf][r * LDF + dof]),
                 (unsigned long long)(kh + (size_t)(kbase + r) * Dh + dof));
    }
#pragma unroll
    for (int j = 0; j < 2; ++j) {
      int c = t + 256 * j;
      int d = c >> 3, tof = (c & 7) * 8; // vst row = d, cols = keys
      async_b128(lds_off(&vst[buf][d * LDF + tof]),
                 (unsigned long long)(vh + (size_t)d * Tseq + kbase + tof));
    }
  };

  const int nkt = 2 * qt + 2;            // causal: keys up to qbase+127
  stage_kv(0, 0);
  asm volatile("s_wait_asynccnt 0x4" ::: "memory");  // Q resident
  __syncthreads();
  bf16x16 qa0 = frag_a(qs + (wv * 16) * LDF, LDF, lane);       // d 0..31
  bf16x16 qa1 = frag_a(qs + (wv * 16) * LDF + 32, LDF, lane);  // d 32..63

  f32x8 o[4];
  float mrow[8], lsum[8];
#pragma unroll
  for (int dt = 0; dt < 4; ++dt)
#pragma unroll
    for (int i = 0; i < 8; ++i) o[dt][i] = 0.f;
#pragma unroll
  for (int v = 0; v < 8; ++v) { mrow[v] = -1e30f; lsum[v] = 0.f; }

  for (int kt = 0; kt < nkt; ++kt) {
    const int kbase = kt * 64;
    const int cur = kt & 1;
    if (kt + 1 < nkt) {
      stage_kv(cur ^ 1, kt + 1);
      asm volatile("s_wait_asynccnt 0x4" ::: "memory");  // batch kt resident
    } else {
      asm volatile("s_wait_asynccnt 0x0" ::: "memory");
    }
    __syncthreads();
    // S = Q K^T  (16 q-rows per wave, 64 keys)
    f32x8 s[4];
#pragma unroll
    for (int nt = 0; nt < 4; ++nt) {
#pragma unroll
      for (int i = 0; i < 8; ++i) s[nt][i] = 0.f;
      s[nt] = wmma_bf16(qa0, frag_b(ks[cur] + (nt * 16) * LDF, LDF, lane), s[nt]);
      s[nt] = wmma_bf16(qa1, frag_b(ks[cur] + (nt * 16) * LDF + 32, LDF, lane), s[nt]);
    }
    if (kbase + 63 > qbase + wv * 16) {   // causal mask (diagonal tiles only)
#pragma unroll
      for (int nt = 0; nt < 4; ++nt)
#pragma unroll
        for (int v = 0; v < 8; ++v) {
          int kg = kbase + nt * 16 + nn;
          int qg = qbase + wv * 16 + v + 8 * gg;
          if (kg > qg) s[nt][v] = -1e30f;
        }
    }
    // online softmax (rows m = v + 8*gg; cols spread across 16-lane group)
    float mnew[8], alpha[8];
#pragma unroll
    for (int v = 0; v < 8; ++v) {
      float mx = fmaxf(fmaxf(s[0][v], s[1][v]), fmaxf(s[2][v], s[3][v]));
#pragma unroll
      for (int off = 1; off < 16; off <<= 1) mx = fmaxf(mx, __shfl_xor(mx, off, 32));
      mnew[v]  = fmaxf(mrow[v], mx);
      alpha[v] = expf(mrow[v] - mnew[v]);
      mrow[v]  = mnew[v];
    }
#pragma unroll
    for (int v = 0; v < 8; ++v) {
      float rs = 0.f;
#pragma unroll
      for (int nt = 0; nt < 4; ++nt) {
        float p = expf(s[nt][v] - mnew[v]); s[nt][v] = p; rs += p;
      }
#pragma unroll
      for (int off = 1; off < 16; off <<= 1) rs += __shfl_xor(rs, off, 32);
      lsum[v] = lsum[v] * alpha[v] + rs;
    }
#pragma unroll
    for (int dt = 0; dt < 4; ++dt)
#pragma unroll
      for (int v = 0; v < 8; ++v) o[dt][v] *= alpha[v];
    {   // P (C-layout) -> per-wave LDS -> A-fragments
      unsigned short* pw = (unsigned short*)(ps + (wv * 16) * LDF);
#pragma unroll
      for (int nt = 0; nt < 4; ++nt)
#pragma unroll
        for (int v = 0; v < 8; ++v) {
          __bf16 hv = (__bf16)s[nt][v];
          pw[(v + 8 * gg) * LDF + nt * 16 + nn] = *(unsigned short*)&hv;
        }
    }
    asm volatile("s_wait_dscnt 0" ::: "memory");  // same-wave LDS RAW
    bf16x16 pa0 = frag_a(ps + (wv * 16) * LDF, LDF, lane);
    bf16x16 pa1 = frag_a(ps + (wv * 16) * LDF + 32, LDF, lane);
#pragma unroll
    for (int dt = 0; dt < 4; ++dt) {
      o[dt] = wmma_bf16(pa0, frag_b(vst[cur] + (dt * 16) * LDF, LDF, lane), o[dt]);
      o[dt] = wmma_bf16(pa1, frag_b(vst[cur] + (dt * 16) * LDF + 32, LDF, lane), o[dt]);
    }
    __syncthreads();   // done reading buf[cur] before it is restaged
  }
  // normalize and store y in [B, T, C] layout (bf16)
#pragma unroll
  for (int v = 0; v < 8; ++v) {
    float inv = 1.f / lsum[v];
    int qg = qbase + wv * 16 + v + 8 * gg;
    __bf16* yr = y + ((size_t)(bb * Tseq + qg)) * Cdim + hh * Dh;
#pragma unroll
    for (int dt = 0; dt < 4; ++dt)
      yr[dt * 16 + nn] = (__bf16)(o[dt][v] * inv);
  }
}

// ---------------------------------------------------------------- launcher
extern "C" void kernel_launch(void* const* d_in, const int* in_sizes, int n_in,
                              void* d_out, int out_size, void* d_ws, size_t ws_size,
                              hipStream_t stream) {
  const float* x       = (const float*)d_in[0];
  const float* ln1_g   = (const float*)d_in[1];
  const float* ln1_b   = (const float*)d_in[2];
  const float* w_attn  = (const float*)d_in[3];
  const float* b_attn  = (const float*)d_in[4];
  const float* w_aproj = (const float*)d_in[5];
  const float* b_aproj = (const float*)d_in[6];
  const float* ln2_g   = (const float*)d_in[7];
  const float* ln2_b   = (const float*)d_in[8];
  const float* w_fc    = (const float*)d_in[9];
  const float* b_fc    = (const float*)d_in[10];
  const float* w_mproj = (const float*)d_in[11];
  const float* b_mproj = (const float*)d_in[12];
  (void)in_sizes; (void)n_in; (void)out_size; (void)ws_size;

  char* base = (char*)d_ws; size_t off = 0;
  auto alloc = [&](size_t bytes) -> void* {
    void* p = base + off;
    off = (off + bytes + 255) & ~(size_t)255;
    return p;
  };
  __bf16* wAttnT  = (__bf16*)alloc((size_t)Cdim * 3 * Cdim * 2);  // [3C][C]
  __bf16* wAprojT = (__bf16*)alloc((size_t)Cdim * Cdim * 2);      // [C][C]
  __bf16* wFcT    = (__bf16*)alloc((size_t)Cdim * 4 * Cdim * 2);  // [4C][C]
  __bf16* wMprojT = (__bf16*)alloc((size_t)4 * Cdim * Cdim * 2);  // [C][4C]
  __bf16* h1      = (__bf16*)alloc((size_t)Mrows * Cdim * 2);
  __bf16* qkvp    = (__bf16*)alloc((size_t)3 * BHTD * 2);
  __bf16* yb      = (__bf16*)alloc((size_t)Mrows * Cdim * 2);
  float*  x1      = (float*) alloc((size_t)Mrows * Cdim * 4);
  __bf16* h2      = (__bf16*)alloc((size_t)Mrows * Cdim * 2);
  __bf16* hmid    = (__bf16*)alloc((size_t)Mrows * 4 * Cdim * 2);

  auto cvtT = [&](const float* s, __bf16* d, int K, int N) {
    cvt_transpose_kernel<<<dim3(N / 32, K / 32), 256, 0, stream>>>(s, d, K, N);
  };
  cvtT(w_attn,  wAttnT,  Cdim, 3 * Cdim);
  cvtT(w_aproj, wAprojT, Cdim, Cdim);
  cvtT(w_fc,    wFcT,    Cdim, 4 * Cdim);
  cvtT(w_mproj, wMprojT, 4 * Cdim, Cdim);

  ln_kernel<<<Mrows, 256, 0, stream>>>(x, ln1_g, ln1_b, h1);

  gemm_bf16_kernel<MODE_QKV><<<dim3(3 * Cdim / 128, Mrows / 128), 256, 0, stream>>>(
      h1, wAttnT, b_attn, nullptr, nullptr, qkvp, 3 * Cdim, Cdim);

  flash_kernel<<<dim3(Bsz * Hh, Tseq / 128), 256, 0, stream>>>(
      qkvp, qkvp + BHTD, qkvp + 2 * (size_t)BHTD, yb);

  gemm_bf16_kernel<MODE_RES><<<dim3(Cdim / 128, Mrows / 128), 256, 0, stream>>>(
      yb, wAprojT, b_aproj, x, x1, nullptr, Cdim, Cdim);

  ln_kernel<<<Mrows, 256, 0, stream>>>(x1, ln2_g, ln2_b, h2);

  gemm_bf16_kernel<MODE_GELU><<<dim3(4 * Cdim / 128, Mrows / 128), 256, 0, stream>>>(
      h2, wFcT, b_fc, nullptr, nullptr, hmid, 4 * Cdim, Cdim);

  gemm_bf16_kernel<MODE_RES><<<dim3(Cdim / 128, Mrows / 128), 256, 0, stream>>>(
      hmid, wMprojT, b_mproj, x1, (float*)d_out, nullptr, Cdim, 4 * Cdim);
}
